// MessagePassingGNN_11811160064083
// MI455X (gfx1250) — compile-verified, matrix operand order checked by reading, not run on hardware
//
#include <hip/hip_runtime.h>

// ---------------------------------------------------------------------------
// MessagePassingGNN on MI455X (gfx1250): fused WMMA (bf16 x bf16 -> f32).
// v2: 32 edges per block (2 M-tiles reuse each B-fragment), persistent bf16
// copy of node features for direct global A-fragment loads, and kernels
// templated on (DIN, DOUT) so all strides/trip-counts are compile-time.
// ---------------------------------------------------------------------------

typedef __attribute__((ext_vector_type(16))) __bf16 v16bf;
typedef __attribute__((ext_vector_type(8)))  __bf16 v8bf;
typedef __attribute__((ext_vector_type(8)))  float  v8f;

#define LN_EPS 1e-5f

// A-fragment (ISA 7.12.2, 16-bit A 16x32): lane<16 -> row=lane, K elems
// {0..7,16..23}; lane>=16 -> row=lane-16, K elems {8..15,24..31}.
// p must already include the per-lane khalf offset.
__device__ __forceinline__ v16bf load_frag(const __bf16* p) {
  v8bf lo = *reinterpret_cast<const v8bf*>(p);
  v8bf hi = *reinterpret_cast<const v8bf*>(p + 16);
  return __builtin_shufflevector(lo, hi, 0,1,2,3,4,5,6,7,
                                 8,9,10,11,12,13,14,15);
}

__device__ __forceinline__ v8f wmma_bf16(v16bf a, v16bf b, v8f c) {
  return __builtin_amdgcn_wmma_f32_16x16x32_bf16(false, a, false, b,
                                                 (short)0, c, false, false);
}

// Convert f32 weight W[K x Nd] into bf16 WMMA-B swizzled layout, K padded to
// Kpad (mult of 32): dst[((kt*NT+nt)*32+lane)*16+e] = W[k][n],
// k = kt*32 + (lane>=16?8:0) + (e&7) + (e>=8?16:0), n = nt*16 + (lane&15).
__global__ void swizzle_weight(const float* __restrict__ W, __bf16* __restrict__ dst,
                               int K, int Nd, int Kpad) {
  int idx = blockIdx.x * blockDim.x + threadIdx.x;
  int total = Kpad * Nd;
  if (idx >= total) return;
  int e    = idx & 15;
  int lane = (idx >> 4) & 31;
  int t    = idx >> 9;
  int NT   = Nd >> 4;
  int nt   = t % NT;
  int kt   = t / NT;
  int n = nt * 16 + (lane & 15);
  int k = kt * 32 + ((lane >> 4) << 3) + (e & 7) + ((e >> 3) << 4);
  float v = (k < K) ? W[(size_t)k * Nd + n] : 0.f;
  dst[idx] = (__bf16)v;
}

__global__ void zero_kernel(float* __restrict__ p, int n) {
  int i = blockIdx.x * blockDim.x + threadIdx.x;
  if (i < n) p[i] = 0.f;
}

// h0 = embed[atom_types], kept in f32 (LN/residual path) and bf16 (WMMA A).
__global__ void embed_kernel(const int* __restrict__ at, const float* __restrict__ emb,
                             float* __restrict__ h, __bf16* __restrict__ hbf) {
  int idx = blockIdx.x * blockDim.x + threadIdx.x;   // 4096*128
  int node = idx >> 7, col = idx & 127;
  float v = emb[at[node] * 128 + col];
  h[idx] = v;
  hbf[idx] = (__bf16)v;
}

__global__ void edge_prep(const int* __restrict__ adj, const int* __restrict__ ebi,
                          const float* __restrict__ coords,
                          int* __restrict__ srcg, int* __restrict__ tgtg,
                          float* __restrict__ ef, int E) {
  int e = blockIdx.x * blockDim.x + threadIdx.x;
  if (e >= E) return;
  int off = ebi[e] * 512;
  int s = adj[2 * e] + off;
  int t = adj[2 * e + 1] + off;
  srcg[e] = s; tgtg[e] = t;
  float bx = coords[3 * t + 0] - coords[3 * s + 0];
  float by = coords[3 * t + 1] - coords[3 * s + 1];
  float bz = coords[3 * t + 2] - coords[3 * s + 2];
  float dist = sqrtf(bx * bx + by * by + bz * bz);
  float cosv = bz / (dist + 1e-8f);
  cosv = fminf(fmaxf(cosv, -1.f + 1e-6f), 1.f - 1e-6f);
  float ang  = acosf(cosv);
  float dih  = sqrtf(bx * bx + by * by);
  float bond = 1.f / (1.f + expf(-2.f * (1.5f - dist)));
  ef[4 * e + 0] = dist; ef[4 * e + 1] = ang;
  ef[4 * e + 2] = dih;  ef[4 * e + 3] = bond;
}

// ---------------------------------------------------------------------------
// Message MLP, 32 edges per block (2 WMMA M-tiles share every B-fragment).
// Stage-1 A comes straight from global bf16 h (src rows, then tgt rows, then
// one LDS-staged 32-wide edge-feature K-tile). m1/m2 ping-pong in one 32KB
// LDS buffer; stage-2 results parked in VGPRs across the barrier.
// ---------------------------------------------------------------------------
template<int DIN, int DOUT>
__global__ __launch_bounds__(256) void msg_kernel(
    const __bf16* __restrict__ hbf,
    const int* __restrict__ srcg, const int* __restrict__ tgtg,
    const float* __restrict__ ef,
    const __bf16* __restrict__ W1, const float* __restrict__ b1,
    const __bf16* __restrict__ W2, const float* __restrict__ b2,
    const __bf16* __restrict__ W3, const float* __restrict__ b3,
    float* __restrict__ agg) {
  constexpr int NT   = DOUT / 16;     // output column tiles
  constexpr int KTH  = DIN / 32;      // K-tiles per h operand
  constexpr int KT2  = DOUT / 32;     // K-tiles for stages 2/3
  constexpr int NJOB = NT / 8;        // nt's per wave (8 waves)
  __shared__ __align__(16) __bf16 sM[32 * DOUT];  // m1, then m2
  __shared__ __align__(16) __bf16 sEF[32 * 32];   // ef K-tile (4 real + pad)
  __shared__ int sS[32], sT[32];
  const int tid = threadIdx.x, lane = tid & 31, wave = tid >> 5;
  const int ebase = blockIdx.x * 32;
  if (tid < 32) { sS[tid] = srcg[ebase + tid]; sT[tid] = tgtg[ebase + tid]; }
  for (int i = tid; i < 32 * 32; i += 256) {
    int r = i >> 5, c = i & 31;
    sEF[i] = (__bf16)(c < 4 ? ef[4 * (ebase + r) + c] : 0.f);
  }
  __syncthreads();
  const int row   = lane & 15;
  const int khalf = (lane >> 4) << 3;
  // Per-lane A base pointers (global bf16 node rows) for both M-tiles.
  const __bf16* aS0 = hbf + (size_t)sS[row]      * DIN + khalf;
  const __bf16* aT0 = hbf + (size_t)sT[row]      * DIN + khalf;
  const __bf16* aS1 = hbf + (size_t)sS[row + 16] * DIN + khalf;
  const __bf16* aT1 = hbf + (size_t)sT[row + 16] * DIN + khalf;
  const v16bf* Bp1 = reinterpret_cast<const v16bf*>(W1) + lane;
  const v16bf* Bp2 = reinterpret_cast<const v16bf*>(W2) + lane;
  const v16bf* Bp3 = reinterpret_cast<const v16bf*>(W3) + lane;

  v8f acc[NJOB][2];
  // ---- stage 1: X=[h_src|h_tgt|ef] @ W1 ----
  for (int j = 0; j < NJOB; ++j) {
    const int nt = wave + j * 8;
    v8f c0 = {}, c1 = {};
    for (int kt = 0; kt < KTH; ++kt) {           // src half
      v16bf b = Bp1[(kt * NT + nt) * 32];
      c0 = wmma_bf16(load_frag(aS0 + kt * 32), b, c0);
      c1 = wmma_bf16(load_frag(aS1 + kt * 32), b, c1);
    }
    for (int kt = 0; kt < KTH; ++kt) {           // tgt half
      v16bf b = Bp1[((KTH + kt) * NT + nt) * 32];
      c0 = wmma_bf16(load_frag(aT0 + kt * 32), b, c0);
      c1 = wmma_bf16(load_frag(aT1 + kt * 32), b, c1);
    }
    {                                            // edge-feature tile
      v16bf b = Bp1[(2 * KTH * NT + nt) * 32];
      c0 = wmma_bf16(load_frag(sEF + row * 32 + khalf), b, c0);
      c1 = wmma_bf16(load_frag(sEF + (row + 16) * 32 + khalf), b, c1);
    }
    acc[j][0] = c0; acc[j][1] = c1;
  }
  for (int j = 0; j < NJOB; ++j) {               // bias + ReLU -> m1
    const int nt = wave + j * 8;
    const int col = nt * 16 + row, r0 = khalf;
    float bias = b1[col];
    for (int e = 0; e < 8; ++e) {
      sM[(r0 + e) * DOUT + col]      = (__bf16)fmaxf(acc[j][0][e] + bias, 0.f);
      sM[(16 + r0 + e) * DOUT + col] = (__bf16)fmaxf(acc[j][1][e] + bias, 0.f);
    }
  }
  __syncthreads();
  // ---- stage 2: m1 @ W2 (results parked in regs) ----
  for (int j = 0; j < NJOB; ++j) {
    const int nt = wave + j * 8;
    v8f c0 = {}, c1 = {};
    for (int kt = 0; kt < KT2; ++kt) {
      v16bf b = Bp2[(kt * NT + nt) * 32];
      c0 = wmma_bf16(load_frag(sM + row * DOUT + kt * 32 + khalf), b, c0);
      c1 = wmma_bf16(load_frag(sM + (row + 16) * DOUT + kt * 32 + khalf), b, c1);
    }
    acc[j][0] = c0; acc[j][1] = c1;
  }
  __syncthreads();                               // all m1 reads complete
  for (int j = 0; j < NJOB; ++j) {               // bias + ReLU -> m2 (same buf)
    const int nt = wave + j * 8;
    const int col = nt * 16 + row, r0 = khalf;
    float bias = b2[col];
    for (int e = 0; e < 8; ++e) {
      sM[(r0 + e) * DOUT + col]      = (__bf16)fmaxf(acc[j][0][e] + bias, 0.f);
      sM[(16 + r0 + e) * DOUT + col] = (__bf16)fmaxf(acc[j][1][e] + bias, 0.f);
    }
  }
  __syncthreads();
  // ---- stage 3: m2 @ W3 -> atomic scatter-add into agg[tgt] ----
  for (int j = 0; j < NJOB; ++j) {
    const int nt = wave + j * 8;
    v8f c0 = {}, c1 = {};
    for (int kt = 0; kt < KT2; ++kt) {
      v16bf b = Bp3[(kt * NT + nt) * 32];
      c0 = wmma_bf16(load_frag(sM + row * DOUT + kt * 32 + khalf), b, c0);
      c1 = wmma_bf16(load_frag(sM + (row + 16) * DOUT + kt * 32 + khalf), b, c1);
    }
    const int col = nt * 16 + row, r0 = khalf;
    float bias = b3[col];
    for (int e = 0; e < 8; ++e) {
      atomicAdd(&agg[(size_t)sT[r0 + e] * DOUT + col],      c0[e] + bias);
      atomicAdd(&agg[(size_t)sT[16 + r0 + e] * DOUT + col], c1[e] + bias);
    }
  }
}

// ---------------------------------------------------------------------------
// Node update per 16-node tile: [h|agg] -> L1 ReLU L2, + residual (WMMA over
// Wres, or identity h), LayerNorm; writes f32 and bf16 copies of new h.
// ---------------------------------------------------------------------------
template<int DIN, int DOUT, int HASRES>
__global__ __launch_bounds__(256) void upd_kernel(
    const float* __restrict__ hin, const __bf16* __restrict__ hinbf,
    const float* __restrict__ agg,
    const __bf16* __restrict__ U1, const float* __restrict__ ub1,
    const __bf16* __restrict__ U2, const float* __restrict__ ub2,
    const __bf16* __restrict__ Rw, const float* __restrict__ rb,
    const float* __restrict__ lng, const float* __restrict__ lnb,
    float* __restrict__ hout, __bf16* __restrict__ houtbf) {
  constexpr int KUP = DIN + DOUT;                 // multiple of 32
  constexpr int NT  = DOUT / 16;
  constexpr int NJOB = NT / 8;
  constexpr size_t RAWB = (size_t)(16 * KUP * 2) > (size_t)(16 * DOUT * 4)
                              ? (size_t)(16 * KUP * 2) : (size_t)(16 * DOUT * 4);
  __shared__ __align__(16) char  sRaw[RAWB];      // X (bf16) then U (f32)
  __shared__ __align__(16) __bf16 sB[16 * DOUT];  // u1
  __bf16* sX = (__bf16*)sRaw;
  float*  sU = (float*)sRaw;
  const int tid = threadIdx.x, lane = tid & 31, wave = tid >> 5;
  const int nbase = blockIdx.x * 16;
  for (int i = tid; i < 16 * KUP; i += 256) {
    int r = i / KUP, c2 = i - r * KUP;
    int node = nbase + r;
    sX[i] = (c2 < DIN) ? hinbf[(size_t)node * DIN + c2]
                       : (__bf16)agg[(size_t)node * DOUT + (c2 - DIN)];
  }
  __syncthreads();
  const int row = lane & 15, khalf = (lane >> 4) << 3;
  for (int j = 0; j < NJOB; ++j) {
    const int nt = wave + j * 8;
    v8f c = {};
    const v16bf* Bp = reinterpret_cast<const v16bf*>(U1) + lane;
    for (int kt = 0; kt < KUP / 32; ++kt)
      c = wmma_bf16(load_frag(sX + row * KUP + kt * 32 + khalf),
                    Bp[(kt * NT + nt) * 32], c);
    const int col = nt * 16 + row, r0 = khalf;
    float bias = ub1[col];
    for (int e = 0; e < 8; ++e)
      sB[(r0 + e) * DOUT + col] = (__bf16)fmaxf(c[e] + bias, 0.f);
  }
  __syncthreads();
  v8f acc[NJOB];
  for (int j = 0; j < NJOB; ++j) {
    const int nt = wave + j * 8;
    v8f c = {};
    if (HASRES) {
      const v16bf* Bp = reinterpret_cast<const v16bf*>(Rw) + lane;
      for (int kt = 0; kt < DIN / 32; ++kt)
        c = wmma_bf16(load_frag(sX + row * KUP + kt * 32 + khalf),
                      Bp[(kt * NT + nt) * 32], c);
    }
    const v16bf* Bp = reinterpret_cast<const v16bf*>(U2) + lane;
    for (int kt = 0; kt < DOUT / 32; ++kt)
      c = wmma_bf16(load_frag(sB + row * DOUT + kt * 32 + khalf),
                    Bp[(kt * NT + nt) * 32], c);
    acc[j] = c;
  }
  __syncthreads();                                // sX/sB reads done
  for (int j = 0; j < NJOB; ++j) {
    const int nt = wave + j * 8;
    const int col = nt * 16 + row, r0 = khalf;
    float bias = ub2[col] + (HASRES ? rb[col] : 0.f);
    for (int e = 0; e < 8; ++e) {
      int node = nbase + r0 + e;
      float v = acc[j][e] + bias;
      if (!HASRES) v += hin[(size_t)node * DIN + col];   // identity residual
      sU[(r0 + e) * DOUT + col] = v;
    }
  }
  __syncthreads();
  for (int rr = 0; rr < 2; ++rr) {                // LayerNorm, 8 waves x 2 rows
    int r = wave * 2 + rr;
    float s = 0.f, s2 = 0.f;
    for (int c2 = lane; c2 < DOUT; c2 += 32) {
      float v = sU[r * DOUT + c2]; s += v; s2 += v * v;
    }
    for (int o = 16; o > 0; o >>= 1) {
      s  += __shfl_xor(s,  o, 32);
      s2 += __shfl_xor(s2, o, 32);
    }
    float mu   = s / (float)DOUT;
    float var  = s2 / (float)DOUT - mu * mu;
    float rstd = rsqrtf(var + LN_EPS);
    int node = nbase + r;
    for (int c2 = lane; c2 < DOUT; c2 += 32) {
      float v = (sU[r * DOUT + c2] - mu) * rstd * lng[c2] + lnb[c2];
      hout[(size_t)node * DOUT + c2]   = v;
      houtbf[(size_t)node * DOUT + c2] = (__bf16)v;
    }
  }
}

// Per-batch mean-pool + 2-layer pool MLP (tiny, scalar).
__global__ void pool_kernel(const float* __restrict__ h,
                            const float* __restrict__ pW1, const float* __restrict__ pb1,
                            const float* __restrict__ pW2, const float* __restrict__ pb2,
                            float* __restrict__ gf) {
  __shared__ float s0[256];
  __shared__ float s1[128];
  int b = blockIdx.x, t = threadIdx.x;
  float acc = 0.f;
  for (int n = 0; n < 512; ++n) acc += h[((size_t)b * 512 + n) * 256 + t];
  s0[t] = acc * (1.f / 512.f);
  __syncthreads();
  if (t < 128) {
    float a = pb1[t];
    for (int k = 0; k < 256; ++k) a += s0[k] * pW1[k * 128 + t];
    s1[t] = fmaxf(a, 0.f);
  }
  __syncthreads();
  if (t < 64) {
    float a = pb2[t];
    for (int k = 0; k < 128; ++k) a += s1[k] * pW2[k * 64 + t];
    gf[b * 64 + t] = a;
  }
}

// Integrate MLP per 16-node tile: [h|gf(b)] (K=320) -> 256 ReLU -> 256 -> out
__global__ __launch_bounds__(256) void integ_kernel(
    const __bf16* __restrict__ hbf, const float* __restrict__ gf,
    const __bf16* __restrict__ I1, const float* __restrict__ ib1,
    const __bf16* __restrict__ I2, const float* __restrict__ ib2,
    float* __restrict__ out) {
  __shared__ __align__(16) __bf16 sX[16 * 320];
  __shared__ __align__(16) __bf16 sB[16 * 256];
  const int tid = threadIdx.x, lane = tid & 31, wave = tid >> 5;
  const int nbase = blockIdx.x * 16;
  for (int i = tid; i < 16 * 320; i += 256) {
    int r = i / 320, c2 = i - r * 320;
    int node = nbase + r;
    sX[i] = (c2 < 256) ? hbf[(size_t)node * 256 + c2]
                       : (__bf16)gf[(node >> 9) * 64 + (c2 - 256)];
  }
  __syncthreads();
  const int row = lane & 15, khalf = (lane >> 4) << 3;
  for (int nt = wave; nt < 16; nt += 8) {
    v8f c = {};
    const v16bf* Bp = reinterpret_cast<const v16bf*>(I1) + lane;
    for (int kt = 0; kt < 10; ++kt)
      c = wmma_bf16(load_frag(sX + row * 320 + kt * 32 + khalf),
                    Bp[(kt * 16 + nt) * 32], c);
    const int col = nt * 16 + row, r0 = khalf;
    float bias = ib1[col];
    for (int e = 0; e < 8; ++e)
      sB[(r0 + e) * 256 + col] = (__bf16)fmaxf(c[e] + bias, 0.f);
  }
  __syncthreads();
  for (int nt = wave; nt < 16; nt += 8) {
    v8f c = {};
    const v16bf* Bp = reinterpret_cast<const v16bf*>(I2) + lane;
    for (int kt = 0; kt < 8; ++kt)
      c = wmma_bf16(load_frag(sB + row * 256 + kt * 32 + khalf),
                    Bp[(kt * 16 + nt) * 32], c);
    const int col = nt * 16 + row, r0 = khalf;
    float bias = ib2[col];
    for (int e = 0; e < 8; ++e)
      out[(size_t)(nbase + r0 + e) * 256 + col] = c[e] + bias;
  }
}

// ---------------------------------------------------------------------------
// Host launcher. Input flattening assumption (setup_inputs() dict insertion
// order, lists/tuples depth-first, layer1's res=None dropped):
//  0 atom_types  1 coordinates  2 adj_list  3 edge_batch_idx  4 embed
//  L0: 5..10 msg(W,b)x3  11..14 upd(W,b)x2  15..16 res(W,b)  17..18 ln(g,b)
//  L1: 19..24 msg        25..28 upd                          29..30 ln
//  L2: 31..36 msg        37..40 upd         41..42 res       43..44 ln
//  pool: 45..48   integ: 49..52
// ---------------------------------------------------------------------------
extern "C" void kernel_launch(void* const* d_in, const int* in_sizes, int n_in,
                              void* d_out, int out_size, void* d_ws, size_t ws_size,
                              hipStream_t stream) {
  const int NODES = 4096, E = 65536;
  const int*   atom_types = (const int*)d_in[0];
  const float* coords     = (const float*)d_in[1];
  const int*   adj        = (const int*)d_in[2];
  const int*   ebi        = (const int*)d_in[3];
  const float* embedW     = (const float*)d_in[4];
  auto F = [&](int i) { return (const float*)d_in[i]; };

  // ---- workspace carve-out ----
  char* ws = (char*)d_ws;
  size_t cur = 0;
  auto alloc = [&](size_t bytes) -> char* {
    char* p = ws + cur;
    cur = (cur + bytes + 255) & ~(size_t)255;
    return p;
  };
  float*  hA    = (float*)alloc((size_t)NODES * 512 * 4);
  float*  hB    = (float*)alloc((size_t)NODES * 512 * 4);
  __bf16* hAb   = (__bf16*)alloc((size_t)NODES * 512 * 2);
  __bf16* hBb   = (__bf16*)alloc((size_t)NODES * 512 * 2);
  float*  agg   = (float*)alloc((size_t)NODES * 512 * 4);
  float*  ef    = (float*)alloc((size_t)E * 4 * 4);
  int*    srcg  = (int*)alloc((size_t)E * 4);
  int*    tgtg  = (int*)alloc((size_t)E * 4);
  float*  gf    = (float*)alloc(8 * 64 * 4);

  auto swz = [&](const float* W, int K, int Nd) -> __bf16* {
    int Kpad = (K + 31) & ~31;
    __bf16* dst = (__bf16*)alloc((size_t)Kpad * Nd * 2);
    int total = Kpad * Nd;
    swizzle_weight<<<(total + 255) / 256, 256, 0, stream>>>(W, dst, K, Nd, Kpad);
    return dst;
  };

  // Layer params + swizzled weights
  // L0: din=128, dout=512 (res W). L1: 512->512 (identity). L2: 512->256 (res W)
  __bf16* m1s[3]; __bf16* m2s[3]; __bf16* m3s[3];
  __bf16* u1s[3]; __bf16* u2s[3]; __bf16* rs[3];
  const int base[3] = {5, 19, 31};
  const int din_[3] = {128, 512, 512}, dout_[3] = {512, 512, 256};
  const int hasres[3] = {1, 0, 1};
  for (int l = 0; l < 3; ++l) {
    int b0 = base[l], din = din_[l], dout = dout_[l];
    m1s[l] = swz(F(b0 + 0), 2 * din + 4, dout);
    m2s[l] = swz(F(b0 + 2), dout, dout);
    m3s[l] = swz(F(b0 + 4), dout, dout);
    u1s[l] = swz(F(b0 + 6), din + dout, dout);
    u2s[l] = swz(F(b0 + 8), dout, dout);
    rs[l]  = hasres[l] ? swz(F(b0 + 10), din, dout) : nullptr;
  }
  const float *pW1 = F(45), *pb1 = F(46), *pW2 = F(47), *pb2 = F(48);
  __bf16* iW1s = swz(F(49), 320, 256);
  const float* ib1 = F(50);
  __bf16* iW2s = swz(F(51), 256, 256);
  const float* ib2 = F(52);

  // ---- forward pass ----
  embed_kernel<<<(NODES * 128) / 256, 256, 0, stream>>>(atom_types, embedW, hA, hAb);
  edge_prep<<<(E + 255) / 256, 256, 0, stream>>>(adj, ebi, coords, srcg, tgtg, ef, E);

  // Layer 0: 128 -> 512 (hA -> hB)
  zero_kernel<<<(NODES * 512 + 255) / 256, 256, 0, stream>>>(agg, NODES * 512);
  msg_kernel<128, 512><<<E / 32, 256, 0, stream>>>(
      hAb, srcg, tgtg, ef, m1s[0], F(6), m2s[0], F(8), m3s[0], F(10), agg);
  upd_kernel<128, 512, 1><<<NODES / 16, 256, 0, stream>>>(
      hA, hAb, agg, u1s[0], F(12), u2s[0], F(14), rs[0], F(16), F(17), F(18),
      hB, hBb);

  // Layer 1: 512 -> 512, identity residual (hB -> hA)
  zero_kernel<<<(NODES * 512 + 255) / 256, 256, 0, stream>>>(agg, NODES * 512);
  msg_kernel<512, 512><<<E / 32, 256, 0, stream>>>(
      hBb, srcg, tgtg, ef, m1s[1], F(20), m2s[1], F(22), m3s[1], F(24), agg);
  upd_kernel<512, 512, 0><<<NODES / 16, 256, 0, stream>>>(
      hB, hBb, agg, u1s[1], F(26), u2s[1], F(28), nullptr, nullptr, F(29), F(30),
      hA, hAb);

  // Layer 2: 512 -> 256 (hA -> hB)
  zero_kernel<<<(NODES * 256 + 255) / 256, 256, 0, stream>>>(agg, NODES * 256);
  msg_kernel<512, 256><<<E / 32, 256, 0, stream>>>(
      hAb, srcg, tgtg, ef, m1s[2], F(32), m2s[2], F(34), m3s[2], F(36), agg);
  upd_kernel<512, 256, 1><<<NODES / 16, 256, 0, stream>>>(
      hA, hAb, agg, u1s[2], F(38), u2s[2], F(40), rs[2], F(42), F(43), F(44),
      hB, hBb);

  pool_kernel<<<8, 256, 0, stream>>>(hB, pW1, pb1, pW2, pb2, gf);
  integ_kernel<<<NODES / 16, 256, 0, stream>>>(hBb, gf, iW1s, ib1, iW2s, ib2,
                                               (float*)d_out);
  (void)in_sizes; (void)n_in; (void)out_size; (void)ws_size;
}
